// HEGNN_85564338471457
// MI455X (gfx1250) — matrix operand order repaired and checked by reading, not compiled, not run on hardware
//
#include <hip/hip_runtime.h>

// ---------------------------------------------------------------------------
// HEGNN forward for MI455X (gfx1250), wave32 + WMMA f32_16x16x32_f16.
//
// Key simplifications proven from the reference:
//   * the coordinate-update branch (_coord / coord_w*) is dead code,
//   * per sub-step k only edges of type k contribute (mask), so we bucket
//     edges by (ht_src,ht_dst) type once and run the edge MLP per type,
//   * dst is naturally sorted -> per-(type,node) padded CSR, no atomics in
//     the hot loop; aggregation fused into the node kernel.
//
// All heavy GEMMs (257x128, 128x128, 256x128, 128x128) run on the matrix
// cores: 16-row M tiles x 128 outputs per wave, K accumulated in 32-chunks
// via __builtin_amdgcn_wmma_f32_16x16x32_f16. Weights are converted once to
// f16, transposed to (n,k) so B fragments are contiguous 32B loads.
// Working set (~34 MB) is L2-resident (192 MB).
// ---------------------------------------------------------------------------

#define N_NODES 4096
#define KNN     11
#define NE      (N_NODES * KNN)   // 45056 edges
#define H       128
#define NLAY    7

typedef __attribute__((ext_vector_type(16))) _Float16 v16h;
typedef __attribute__((ext_vector_type(8)))  _Float16 v8h;
typedef __attribute__((ext_vector_type(8)))  float    v8f;

__device__ __forceinline__ float silu_f(float x) { return x / (1.0f + __expf(-x)); }

// A fragment (16x32 f16, M x K). Lane l (l<16) = row M=l, K=kbase+0..15;
// lane l+16 = row M=l, K=kbase+16..31.
__device__ __forceinline__ v16h load_a_frag(const _Float16* row, int kbase, int hi) {
  return *reinterpret_cast<const v16h*>(row + kbase + hi * 16);
}

// B fragment (32x16 f16, K x N), weights stored transposed (n,k) row-major:
// lane n (n<16) = column N=n, K = kbase+0..15 ; lane n+16 = K = kbase+16..31.
__device__ __forceinline__ v16h load_b_frag(const _Float16* wrow_n, int kbase, int hi) {
  return *reinterpret_cast<const v16h*>(wrow_n + kbase + hi * 16);
}

// ------------------------------ prep kernels -------------------------------

__global__ __launch_bounds__(256) void compute_ht_kernel(
    const float* __restrict__ hit_type, int* __restrict__ ht) {
  int d = blockIdx.x * blockDim.x + threadIdx.x;
  if (d >= N_NODES) return;
  const float* r = hit_type + (size_t)d * 4;
  int best = 0; float bv = r[0];
#pragma unroll
  for (int c = 1; c < 4; ++c) { if (r[c] > bv) { bv = r[c]; best = c; } }
  ht[d] = best;
}

// one wave per node: brute-force 11-NN (same distance formula as reference)
__global__ __launch_bounds__(256) void knn_kernel(
    const float* __restrict__ node_h, int* __restrict__ nbr) {
  __shared__ float s_d[8][32 * KNN];
  __shared__ int   s_i[8][32 * KNN];
  int wid = threadIdx.x >> 5, lane = threadIdx.x & 31;
  int d = blockIdx.x * 8 + wid;
  float x0 = node_h[(size_t)d * 9 + 0];
  float x1 = node_h[(size_t)d * 9 + 1];
  float x2 = node_h[(size_t)d * 9 + 2];
  float x2d = x0 * x0 + x1 * x1 + x2 * x2;
  float bd[KNN]; int bi[KNN];
#pragma unroll
  for (int s = 0; s < KNN; ++s) { bd[s] = 3.4e38f; bi[s] = 0; }
  for (int j = lane; j < N_NODES; j += 32) {
    float y0 = node_h[(size_t)j * 9 + 0];
    float y1 = node_h[(size_t)j * 9 + 1];
    float y2 = node_h[(size_t)j * 9 + 2];
    float dot = x0 * y0 + x1 * y1 + x2 * y2;
    float x2j = y0 * y0 + y1 * y1 + y2 * y2;
    float dist = x2d + x2j - 2.0f * dot;
    if (j == d) dist = 3.4e38f;
    if (dist < bd[KNN - 1]) {
      int p = KNN - 1;
      while (p > 0 && bd[p - 1] > dist) { bd[p] = bd[p - 1]; bi[p] = bi[p - 1]; --p; }
      bd[p] = dist; bi[p] = j;
    }
  }
#pragma unroll
  for (int s = 0; s < KNN; ++s) { s_d[wid][lane * KNN + s] = bd[s]; s_i[wid][lane * KNN + s] = bi[s]; }
  __syncthreads();
  if (lane == 0) {
    for (int s = 0; s < KNN; ++s) {
      float best = 3.4e38f; int bq = 0;
      for (int q = 0; q < 32 * KNN; ++q)
        if (s_d[wid][q] < best) { best = s_d[wid][q]; bq = q; }
      nbr[(size_t)d * KNN + s] = s_i[wid][bq];
      s_d[wid][bq] = 3.4e38f;
    }
  }
}

// bucket edges by type, build padded CSR + compact lists, radial
__global__ __launch_bounds__(256) void edge_build_kernel(
    const float* __restrict__ node_h, const int* __restrict__ ht,
    const int* __restrict__ nbr, float* __restrict__ radial,
    int* __restrict__ cnt /*[4][N]*/, int* __restrict__ slot /*[4][N*K]*/,
    int* __restrict__ csrc /*[4][NE]*/, int* __restrict__ ce /*[4][NE]*/,
    int* __restrict__ type_count /*[4]*/) {
  int d = blockIdx.x * blockDim.x + threadIdx.x;
  if (d >= N_NODES) return;
  float x0 = node_h[(size_t)d * 9 + 0];
  float x1 = node_h[(size_t)d * 9 + 1];
  float x2 = node_h[(size_t)d * 9 + 2];
  int htd = ht[d];
  int local[4] = {0, 0, 0, 0};
  for (int j = 0; j < KNN; ++j) {
    int s = nbr[(size_t)d * KNN + j];
    int e = d * KNN + j;
    float d0 = node_h[(size_t)s * 9 + 0] - x0;
    float d1 = node_h[(size_t)s * 9 + 1] - x1;
    float d2 = node_h[(size_t)s * 9 + 2] - x2;
    radial[e] = d0 * d0 + d1 * d1 + d2 * d2;
    int t = ((ht[s] == 3) ? 2 : 0) + ((htd == 3) ? 1 : 0); // (2,2)(2,3)(3,2)(3,3)
    int ci = atomicAdd(&type_count[t], 1);
    csrc[t * NE + ci] = s;
    ce[t * NE + ci]   = e;
    slot[(size_t)t * (N_NODES * KNN) + (size_t)d * KNN + local[t]] = ci;
    local[t]++;
  }
#pragma unroll
  for (int t = 0; t < 4; ++t) cnt[t * N_NODES + d] = local[t];
}

// convert all used weights to f16, transposed (n,k)
__global__ __launch_bounds__(256) void convert_weights_kernel(
    const float* __restrict__ eW1, const float* __restrict__ eW2,
    const float* __restrict__ nW1, const float* __restrict__ nW2,
    _Float16* __restrict__ w1t, float* __restrict__ w1r,
    _Float16* __restrict__ w2t, _Float16* __restrict__ n1t,
    _Float16* __restrict__ n2t) {
  const int LK = NLAY * 4;
  int stride = gridDim.x * blockDim.x;
  int tid = blockIdx.x * blockDim.x + threadIdx.x;
  // edge_w1: (LK,257,128) -> w1t[lk][n][kk] = W1[lk][kk+1][n]; w1r[lk][n] = W1[lk][0][n]
  for (int i = tid; i < LK * 128 * 256; i += stride) {
    int lk = i / (128 * 256); int n = (i / 256) & 127; int kk = i & 255;
    w1t[i] = (_Float16)eW1[((size_t)lk * 257 + (kk + 1)) * 128 + n];
  }
  for (int i = tid; i < LK * 128; i += stride) {
    int lk = i / 128; int n = i & 127;
    w1r[i] = eW1[((size_t)lk * 257) * 128 + n];
  }
  // edge_w2: (LK,128,128) -> w2t[lk][n][k] = W2[lk][k][n]
  for (int i = tid; i < LK * 128 * 128; i += stride) {
    int lk = i / (128 * 128); int n = (i / 128) & 127; int k = i & 127;
    w2t[i] = (_Float16)eW2[((size_t)lk * 128 + k) * 128 + n];
  }
  // node_w1: (LK,256,128) -> n1t[lk][n][kk] = NW1[lk][kk][n]
  for (int i = tid; i < LK * 128 * 256; i += stride) {
    int lk = i / (128 * 256); int n = (i / 256) & 127; int kk = i & 255;
    n1t[i] = (_Float16)nW1[((size_t)lk * 256 + kk) * 128 + n];
  }
  // node_w2
  for (int i = tid; i < LK * 128 * 128; i += stride) {
    int lk = i / (128 * 128); int n = (i / 128) & 127; int k = i & 127;
    n2t[i] = (_Float16)nW2[((size_t)lk * 128 + k) * 128 + n];
  }
}

__global__ __launch_bounds__(256) void embed_in_kernel(
    const float* __restrict__ node_h, const int* __restrict__ ht,
    const float* __restrict__ w /*(2,6,128)*/, const float* __restrict__ b /*(2,128)*/,
    float* __restrict__ hh, _Float16* hh16) {
  int idx = blockIdx.x * blockDim.x + threadIdx.x;
  if (idx >= N_NODES * H) return;
  int node = idx >> 7, h = idx & 127;
  int sel = (ht[node] == 2) ? 0 : 1;
  float s = b[sel * H + h];
#pragma unroll
  for (int f = 0; f < 6; ++f)
    s += node_h[(size_t)node * 9 + 3 + f] * w[((size_t)sel * 6 + f) * H + h];
  hh[idx] = s;
  hh16[idx] = (_Float16)s;
}

// ------------------------------ edge MLP (WMMA) ----------------------------
// One wave per 16-edge tile; 8 waves per block. Computes
//   ef = SILU(SILU([radial|hh_src|hh_dst] @ W1 + b1) @ W2 + b2)
// and writes per-edge ef (compact index) to ef_buf.
__global__ __launch_bounds__(256) void edge_mlp_kernel(
    const _Float16* __restrict__ hh16,
    const int* __restrict__ csrc, const int* __restrict__ ce,
    const int* __restrict__ type_count, const float* __restrict__ radial,
    const _Float16* __restrict__ w1t /*[128][256]*/, const float* __restrict__ w1r,
    const float* __restrict__ b1,
    const _Float16* __restrict__ w2t /*[128][128]*/, const float* __restrict__ b2,
    float* __restrict__ ef_buf /*[NE][128]*/) {
  __shared__ __align__(32) _Float16 s_ef1[8][16 * H];
  __shared__ float s_rad[8][16];
  int wid = threadIdx.x >> 5, lane = threadIdx.x & 31;
  int tile = blockIdx.x * 8 + wid;
  int cnt = type_count[0];
  bool tile_active = (tile * 16) < cnt;
  int m = lane & 15, hi = lane >> 4, nl = lane & 15;

  int src = 0, e = 0; float rad_m = 0.0f;
  if (tile_active) {
    int gi = tile * 16 + m;
    if (gi > cnt - 1) gi = cnt - 1;
    src = csrc[gi];
    e   = ce[gi];
    rad_m = radial[e];
  }
  int dstn = e / KNN;
  if (lane < 16) s_rad[wid][m] = rad_m;

  const _Float16* srow = hh16 + (size_t)src * H;
  const _Float16* drow = hh16 + (size_t)dstn * H;
  v16h A[8];
#pragma unroll
  for (int c = 0; c < 4; ++c) A[c] = load_a_frag(srow, c * 32, hi);        // K 0..127 = hh[src]
#pragma unroll
  for (int c = 0; c < 4; ++c) A[4 + c] = load_a_frag(drow, c * 32, hi);    // K 128..255 = hh[dst]
  __builtin_prefetch(w1t, 0, 3);
  __syncthreads();

  // layer 1: K=256 (+ radial rank-1 in epilogue)
  for (int nt = 0; nt < 8; ++nt) {
    int n = nt * 16 + nl;
    const _Float16* wrow = w1t + (size_t)n * 256;
    v8f acc = {};
#pragma unroll
    for (int c = 0; c < 8; ++c) {
      v16h B = load_b_frag(wrow, c * 32, hi);
      acc = __builtin_amdgcn_wmma_f32_16x16x32_f16(false, A[c], false, B,
                                                   (short)0, acc, false, false);
    }
    float bias = b1[n], wr = w1r[n];
#pragma unroll
    for (int r = 0; r < 8; ++r) {
      int mr = r + hi * 8;
      float v = acc[r] + bias + s_rad[wid][mr] * wr;
      s_ef1[wid][mr * H + n] = (_Float16)silu_f(v);
    }
  }
  __syncthreads();

  // layer 2: K=128 from LDS
  v16h A2[4];
  const _Float16* erow = &s_ef1[wid][m * H];
#pragma unroll
  for (int c = 0; c < 4; ++c) A2[c] = load_a_frag(erow, c * 32, hi);
  for (int nt = 0; nt < 8; ++nt) {
    int n = nt * 16 + nl;
    const _Float16* wrow = w2t + (size_t)n * H;
    v8f acc = {};
#pragma unroll
    for (int c = 0; c < 4; ++c) {
      v16h B = load_b_frag(wrow, c * 32, hi);
      acc = __builtin_amdgcn_wmma_f32_16x16x32_f16(false, A2[c], false, B,
                                                   (short)0, acc, false, false);
    }
    float bias = b2[n];
#pragma unroll
    for (int r = 0; r < 8; ++r) {
      int mr = r + hi * 8;
      int gi = tile * 16 + mr;
      if (gi < cnt) ef_buf[(size_t)gi * H + n] = silu_f(acc[r] + bias);
    }
  }
}

// ------------------------------ node MLP (WMMA) ----------------------------
// One wave per 16-node tile; 4 waves per block. Fuses ef aggregation,
//   dh = SILU([hh|ef_sum] @ NW1 + nb1) @ NW2 + nb2
//   hh = (2*hh + recv*dh) * out_scale   (out_scale = 0.25 on sub-step 3)
__global__ __launch_bounds__(128) void node_mlp_kernel(
    _Float16* hh16, float* __restrict__ hh,
    const float* __restrict__ ef_buf,
    const int* __restrict__ slot /*[N][K]*/, const int* __restrict__ cntv /*[N]*/,
    const _Float16* __restrict__ n1t /*[128][256]*/, const float* __restrict__ nb1,
    const _Float16* __restrict__ n2t /*[128][128]*/, const float* __restrict__ nb2,
    float out_scale) {
  __shared__ __align__(32) _Float16 s_ef[4][16 * H];
  __shared__ __align__(32) _Float16 s_h1[4][16 * H];
  int wid = threadIdx.x >> 5, lane = threadIdx.x & 31;
  int tile = blockIdx.x * 4 + wid;
  int node0 = tile * 16;

  // aggregate ef_sum for this tile -> f16 in LDS
  for (int idx = lane; idx < 16 * H; idx += 32) {
    int mr = idx >> 7, col = idx & 127;
    int node = node0 + mr;
    int c = cntv[node];
    float s = 0.0f;
    for (int j = 0; j < c; ++j) {
      int ci = slot[(size_t)node * KNN + j];
      s += ef_buf[(size_t)ci * H + col];
    }
    s_ef[wid][mr * H + col] = (_Float16)s;
  }
  __syncthreads();

  int m = lane & 15, hi = lane >> 4, nl = lane & 15;
  const _Float16* hrow = hh16 + (size_t)(node0 + m) * H;
  const _Float16* erow = &s_ef[wid][m * H];
  v16h A[8];
#pragma unroll
  for (int c = 0; c < 4; ++c) A[c] = load_a_frag(hrow, c * 32, hi);        // K 0..127 = hh
#pragma unroll
  for (int c = 0; c < 4; ++c) A[4 + c] = load_a_frag(erow, c * 32, hi);    // K 128..255 = ef_sum

  // layer 1: K=256
  for (int nt = 0; nt < 8; ++nt) {
    int n = nt * 16 + nl;
    const _Float16* wrow = n1t + (size_t)n * 256;
    v8f acc = {};
#pragma unroll
    for (int c = 0; c < 8; ++c) {
      v16h B = load_b_frag(wrow, c * 32, hi);
      acc = __builtin_amdgcn_wmma_f32_16x16x32_f16(false, A[c], false, B,
                                                   (short)0, acc, false, false);
    }
    float bias = nb1[n];
#pragma unroll
    for (int r = 0; r < 8; ++r) {
      int mr = r + hi * 8;
      s_h1[wid][mr * H + n] = (_Float16)silu_f(acc[r] + bias);
    }
  }
  __syncthreads();

  // layer 2: K=128, fused hh update
  v16h A2[4];
  const _Float16* h1row = &s_h1[wid][m * H];
#pragma unroll
  for (int c = 0; c < 4; ++c) A2[c] = load_a_frag(h1row, c * 32, hi);
  for (int nt = 0; nt < 8; ++nt) {
    int n = nt * 16 + nl;
    const _Float16* wrow = n2t + (size_t)n * H;
    v8f acc = {};
#pragma unroll
    for (int c = 0; c < 4; ++c) {
      v16h B = load_b_frag(wrow, c * 32, hi);
      acc = __builtin_amdgcn_wmma_f32_16x16x32_f16(false, A2[c], false, B,
                                                   (short)0, acc, false, false);
    }
    float bias = nb2[n];
#pragma unroll
    for (int r = 0; r < 8; ++r) {
      int mr = r + hi * 8;
      int node = node0 + mr;
      float dh = acc[r] + bias;
      float rv = (cntv[node] > 0) ? 1.0f : 0.0f;
      size_t o = (size_t)node * H + n;
      float hn = (2.0f * hh[o] + rv * dh) * out_scale;
      hh[o] = hn;
      hh16[o] = (_Float16)hn;
    }
  }
}

__global__ __launch_bounds__(256) void embed_out_kernel(
    const float* __restrict__ hh, const float* __restrict__ node_h,
    const int* __restrict__ ht, const float* __restrict__ w /*(2,131,4)*/,
    const float* __restrict__ b /*(2,4)*/, float* __restrict__ out) {
  int node = blockIdx.x * blockDim.x + threadIdx.x;
  if (node >= N_NODES) return;
  int sel = (ht[node] == 2) ? 0 : 1;
  const float* W = w + (size_t)sel * 131 * 4;
  float o[4];
#pragma unroll
  for (int c = 0; c < 4; ++c) o[c] = b[sel * 4 + c];
  for (int j = 0; j < H; ++j) {
    float v = hh[(size_t)node * H + j];
#pragma unroll
    for (int c = 0; c < 4; ++c) o[c] += v * W[j * 4 + c];
  }
#pragma unroll
  for (int jj = 0; jj < 3; ++jj) {
    float v = node_h[(size_t)node * 9 + jj];
#pragma unroll
    for (int c = 0; c < 4; ++c) o[c] += v * W[(H + jj) * 4 + c];
  }
#pragma unroll
  for (int c = 0; c < 4; ++c) out[(size_t)node * 4 + c] = o[c];
}

// ------------------------------- launch ------------------------------------

extern "C" void kernel_launch(void* const* d_in, const int* in_sizes, int n_in,
                              void* d_out, int out_size, void* d_ws, size_t ws_size,
                              hipStream_t stream) {
  const float* node_h    = (const float*)d_in[0];
  const float* hit_type  = (const float*)d_in[1];
  const float* emb_in_w  = (const float*)d_in[2];
  const float* emb_in_b  = (const float*)d_in[3];
  const float* emb_out_w = (const float*)d_in[4];
  const float* emb_out_b = (const float*)d_in[5];
  const float* edge_w1   = (const float*)d_in[6];
  const float* edge_b1   = (const float*)d_in[7];
  const float* edge_w2   = (const float*)d_in[8];
  const float* edge_b2   = (const float*)d_in[9];
  // d_in[10..12] = coord_w1/coord_b1/coord_w2 -> dead code in reference, unused
  const float* node_w1   = (const float*)d_in[13];
  const float* node_b1   = (const float*)d_in[14];
  const float* node_w2   = (const float*)d_in[15];
  const float* node_b2   = (const float*)d_in[16];

  char* ws = (char*)d_ws;
  size_t off = 0;
  auto alloc = [&](size_t bytes) -> void* {
    void* p = (void*)(ws + off);
    off += (bytes + 255) & ~(size_t)255;
    return p;
  };
  // total ~34 MB
  int*      ht_buf     = (int*)alloc((size_t)N_NODES * 4);
  int*      nbr        = (int*)alloc((size_t)NE * 4);
  float*    radial     = (float*)alloc((size_t)NE * 4);
  int*      cnt        = (int*)alloc((size_t)4 * N_NODES * 4);
  int*      slot       = (int*)alloc((size_t)4 * NE * 4);
  int*      csrc       = (int*)alloc((size_t)4 * NE * 4);
  int*      ce         = (int*)alloc((size_t)4 * NE * 4);
  int*      type_count = (int*)alloc(16);
  float*    hh         = (float*)alloc((size_t)N_NODES * H * 4);
  _Float16* hh16       = (_Float16*)alloc((size_t)N_NODES * H * 2);
  float*    ef_buf     = (float*)alloc((size_t)NE * H * 4);
  _Float16* w1t        = (_Float16*)alloc((size_t)NLAY * 4 * 128 * 256 * 2);
  float*    w1r        = (float*)alloc((size_t)NLAY * 4 * 128 * 4);
  _Float16* w2t        = (_Float16*)alloc((size_t)NLAY * 4 * 128 * 128 * 2);
  _Float16* n1t        = (_Float16*)alloc((size_t)NLAY * 4 * 128 * 256 * 2);
  _Float16* n2t        = (_Float16*)alloc((size_t)NLAY * 4 * 128 * 128 * 2);
  (void)ws_size; (void)in_sizes; (void)n_in; (void)out_size;

  hipMemsetAsync(type_count, 0, 16, stream);
  compute_ht_kernel<<<(N_NODES + 255) / 256, 256, 0, stream>>>(hit_type, ht_buf);
  knn_kernel<<<N_NODES / 8, 256, 0, stream>>>(node_h, nbr);
  edge_build_kernel<<<(N_NODES + 255) / 256, 256, 0, stream>>>(
      node_h, ht_buf, nbr, radial, cnt, slot, csrc, ce, type_count);
  convert_weights_kernel<<<1024, 256, 0, stream>>>(
      edge_w1, edge_w2, node_w1, node_w2, w1t, w1r, w2t, n1t, n2t);
  embed_in_kernel<<<(N_NODES * H + 255) / 256, 256, 0, stream>>>(
      node_h, ht_buf, emb_in_w, emb_in_b, hh, hh16);

  const int edge_blocks = ((NE + 15) / 16 + 7) / 8;      // worst case: all edges one type
  const int node_blocks = (N_NODES / 16) / 4;
  for (int l = 0; l < NLAY; ++l) {
    for (int k = 0; k < 4; ++k) {
      int lk = l * 4 + k;
      edge_mlp_kernel<<<edge_blocks, 256, 0, stream>>>(
          hh16, csrc + (size_t)k * NE, ce + (size_t)k * NE, type_count + k, radial,
          w1t + (size_t)lk * 128 * 256, w1r + (size_t)lk * 128, edge_b1 + (size_t)lk * 128,
          w2t + (size_t)lk * 128 * 128, edge_b2 + (size_t)lk * 128, ef_buf);
      float out_scale = (k == 3) ? 0.25f : 1.0f;         // hh /= 4 at layer end
      node_mlp_kernel<<<node_blocks, 128, 0, stream>>>(
          hh16, hh, ef_buf, slot + (size_t)k * NE, cnt + (size_t)k * N_NODES,
          n1t + (size_t)lk * 128 * 256, node_b1 + (size_t)lk * 128,
          n2t + (size_t)lk * 128 * 128, node_b2 + (size_t)lk * 128, out_scale);
    }
  }
  embed_out_kernel<<<(N_NODES + 255) / 256, 256, 0, stream>>>(
      hh, node_h, ht_buf, emb_out_w, emb_out_b, (float*)d_out);
}
